// MyBinaryDense_91070486544907
// MI455X (gfx1250) — compile-verified, matrix operand order checked by reading, not run on hardware
//
#include <hip/hip_runtime.h>

// Binary-weight dense layer: out = relu(x @ sign(W) + b)
// N=8192, CIN=4096, NU=4096. Compute-bound on MI455X (AI ~820 FLOP/B vs
// machine balance ~200), so drive the bf16 WMMA pipe with split-precision
// (x = x_hi + x_lo in bf16, W binarized to exact bf16 +/-1), f32 accumulate.

typedef __attribute__((ext_vector_type(16))) __bf16 v16bf;
typedef __attribute__((ext_vector_type(8)))  __bf16 v8bf;
typedef __attribute__((ext_vector_type(8)))  float  v8f;

#define NROWS 8192
#define CIN   4096
#define NUNIT 4096
#define BM    128
#define BN    128
#define BK    32
#define KT    (CIN / BK)

union FragAB {
    v16bf v;
    v8bf  h[2];
};

__device__ __forceinline__ void split_bf16(float x, unsigned short& hi, unsigned short& lo) {
    unsigned u = __float_as_uint(x);
    hi = (unsigned short)(u >> 16);                 // truncated bf16 (exact prefix)
    float xh = __uint_as_float(u & 0xFFFF0000u);
    float r  = x - xh;                              // residual, |r| < 2^-7 |x|
    unsigned ur = __float_as_uint(r);
    unsigned rr = ur + 0x7FFFu + ((ur >> 16) & 1u); // RNE to bf16
    lo = (unsigned short)(rr >> 16);
}

__global__ __launch_bounds__(256) void binary_dense_wmma(
    const float* __restrict__ x, const float* __restrict__ W,
    const float* __restrict__ bias, float* __restrict__ out)
{
    __shared__ __align__(16) unsigned short ldsAhi[BM * BK]; // x high bf16, [M][K]
    __shared__ __align__(16) unsigned short ldsAlo[BM * BK]; // x low  bf16, [M][K]
    __shared__ __align__(16) unsigned short ldsB  [BN * BK]; // sign(W),     [N][K]

    const int tid   = threadIdx.x;
    const int lane  = tid & 31;
    const int wave  = tid >> 5;
    const int waveM = wave & 3;   // 4 row-bands of 32
    const int waveN = wave >> 2;  // 2 col-bands of 64

    const int mBlock = blockIdx.y * BM;
    const int nBlock = blockIdx.x * BN;

    // global-load mapping (256 threads)
    const int xr  = tid >> 3;        // x tile: base row (0..31), 4 rows/thread
    const int xc4 = (tid & 7) * 4;   // x tile: float col
    const int wk  = tid >> 5;        // W tile: base k (0..7), 4 k/thread
    const int wc4 = (tid & 31) * 4;  // W tile: float col

    float4 xreg[4];
    float4 wreg[4];

    auto load_tile = [&](int kt) {
        const float* xp = x + (size_t)mBlock * CIN + (size_t)kt * BK;
#pragma unroll
        for (int i = 0; i < 4; ++i)
            xreg[i] = *reinterpret_cast<const float4*>(xp + (size_t)(xr + 32 * i) * CIN + xc4);
        const float* wp = W + (size_t)(kt * BK) * NUNIT + nBlock;
#pragma unroll
        for (int i = 0; i < 4; ++i)
            wreg[i] = *reinterpret_cast<const float4*>(wp + (size_t)(wk + 8 * i) * NUNIT + wc4);
    };

    auto store_tile = [&]() {
#pragma unroll
        for (int i = 0; i < 4; ++i) {
            float v[4] = {xreg[i].x, xreg[i].y, xreg[i].z, xreg[i].w};
            unsigned short h[4], l[4];
#pragma unroll
            for (int j = 0; j < 4; ++j) split_bf16(v[j], h[j], l[j]);
            const int row = xr + 32 * i;
            uint2 ph, pl;
            ph.x = (unsigned)h[0] | ((unsigned)h[1] << 16);
            ph.y = (unsigned)h[2] | ((unsigned)h[3] << 16);
            pl.x = (unsigned)l[0] | ((unsigned)l[1] << 16);
            pl.y = (unsigned)l[2] | ((unsigned)l[3] << 16);
            *reinterpret_cast<uint2*>(&ldsAhi[row * BK + xc4]) = ph;
            *reinterpret_cast<uint2*>(&ldsAlo[row * BK + xc4]) = pl;
        }
#pragma unroll
        for (int i = 0; i < 4; ++i) {
            float v[4] = {wreg[i].x, wreg[i].y, wreg[i].z, wreg[i].w};
            const int k = wk + 8 * i;
#pragma unroll
            for (int j = 0; j < 4; ++j) {
                // sign-binarize: +/-1.0 bf16 = 0x3F80 | signbit
                unsigned short s =
                    (unsigned short)(0x3F80u | ((__float_as_uint(v[j]) >> 16) & 0x8000u));
                ldsB[(wc4 + j) * BK + k] = s;  // transpose to [N][K]
            }
        }
    };

    v8f acc[2][4];
#pragma unroll
    for (int mi = 0; mi < 2; ++mi)
#pragma unroll
        for (int ni = 0; ni < 4; ++ni)
#pragma unroll
            for (int e = 0; e < 8; ++e) acc[mi][ni][e] = 0.0f;

    const int mrow = lane & 15;
    const int kb   = (lane < 16) ? 0 : 8;  // wave32 WMMA 16-bit operand K layout

    load_tile(0);
    for (int kt = 0; kt < KT; ++kt) {
        if (kt) __syncthreads();
        store_tile();
        __syncthreads();
        if (kt + 1 < KT) load_tile(kt + 1);  // prefetch overlaps WMMA issue

        FragAB afr[2][2], bfr[4];
#pragma unroll
        for (int mi = 0; mi < 2; ++mi) {
            const int row = waveM * 32 + mi * 16 + mrow;
            afr[mi][0].h[0] = *reinterpret_cast<const v8bf*>(&ldsAhi[row * BK + kb]);
            afr[mi][0].h[1] = *reinterpret_cast<const v8bf*>(&ldsAhi[row * BK + kb + 16]);
            afr[mi][1].h[0] = *reinterpret_cast<const v8bf*>(&ldsAlo[row * BK + kb]);
            afr[mi][1].h[1] = *reinterpret_cast<const v8bf*>(&ldsAlo[row * BK + kb + 16]);
        }
#pragma unroll
        for (int ni = 0; ni < 4; ++ni) {
            const int col = waveN * 64 + ni * 16 + mrow;
            bfr[ni].h[0] = *reinterpret_cast<const v8bf*>(&ldsB[col * BK + kb]);
            bfr[ni].h[1] = *reinterpret_cast<const v8bf*>(&ldsB[col * BK + kb + 16]);
        }
#pragma unroll
        for (int mi = 0; mi < 2; ++mi)
#pragma unroll
            for (int ni = 0; ni < 4; ++ni) {
                acc[mi][ni] = __builtin_amdgcn_wmma_f32_16x16x32_bf16(
                    false, afr[mi][0].v, false, bfr[ni].v, (short)0, acc[mi][ni], false, false);
                acc[mi][ni] = __builtin_amdgcn_wmma_f32_16x16x32_bf16(
                    false, afr[mi][1].v, false, bfr[ni].v, (short)0, acc[mi][ni], false, false);
            }
    }

    // Epilogue: bias + relu, C/D layout: lane%16 = N col, VGPR r = row (+8 for upper half-wave)
#pragma unroll
    for (int mi = 0; mi < 2; ++mi)
#pragma unroll
        for (int ni = 0; ni < 4; ++ni) {
            const int col   = nBlock + waveN * 64 + ni * 16 + (lane & 15);
            const float bv  = bias[col];
            const int rbase = mBlock + waveM * 32 + mi * 16 + ((lane < 16) ? 0 : 8);
#pragma unroll
            for (int r = 0; r < 8; ++r) {
                float v = acc[mi][ni][r] + bv;
                out[(size_t)(rbase + r) * NUNIT + col] = fmaxf(v, 0.0f);
            }
        }
}

extern "C" void kernel_launch(void* const* d_in, const int* in_sizes, int n_in,
                              void* d_out, int out_size, void* d_ws, size_t ws_size,
                              hipStream_t stream) {
    (void)in_sizes; (void)n_in; (void)out_size; (void)d_ws; (void)ws_size;
    const float* x = (const float*)d_in[0];
    const float* W = (const float*)d_in[1];
    const float* b = (const float*)d_in[2];
    float* out = (float*)d_out;

    dim3 grid(NUNIT / BN, NROWS / BM);  // 32 x 64 blocks
    binary_dense_wmma<<<grid, 256, 0, stream>>>(x, W, b, out);
}
